// QueryScan_MultiscaleEncoder_62079457296621
// MI455X (gfx1250) — compile-verified, matrix operand order checked by reading, not run on hardware
//
#include <hip/hip_runtime.h>
#include <hip/hip_bf16.h>

// ---------------- constants from the reference ----------------
#define D      256
#define NQ     25
#define T      3
#define HW0    1024
#define HW1    256
#define HW2    64
#define K0     256
#define K1     64
#define K2     16
#define LSEG   340          // per-t segment: 256 + 1 + 64 + 1 + 16 + 1 + 1
#define LTOT   1020         // T * LSEG
#define NTOK   25500        // NQ * LTOT
#define NTOKP  25504        // padded to multiple of 32 for WMMA tiling
#define DFF    1024
#define HWALL  1344         // 1024+256+64
#define ROWS   4032         // T * HWALL

typedef __attribute__((ext_vector_type(16))) _Float16 v16h;
typedef __attribute__((ext_vector_type(8)))  _Float16 v8h;
typedef __attribute__((ext_vector_type(8)))  float    v8f;

// ---------------- register-blocked WMMA GEMM --------------------------------
// C = act(A(MxK,f16) * Bt(NxK,f16)^T + bias); one wave computes MT x NT tiles
// of 16x16 (A fragments reused across NT columns, 8 WMMAs per K-step for 2x4).
template <int MT, int NT>
__global__ void wmma_gemm_t(const _Float16* __restrict__ A,
                            const _Float16* __restrict__ Bt,
                            const float* __restrict__ bias,
                            float* __restrict__ Cf, _Float16* __restrict__ Ch,
                            int M, int N, int K, int relu) {
  const int lane = threadIdx.x;                 // 0..31 (wave32)
  const int l15  = lane & 15;
  const int kb   = (lane < 16) ? 0 : 8;         // ISA: half-wave K sub-offset
  const int rm0  = blockIdx.x * (16 * MT);
  const int cn0  = blockIdx.y * (16 * NT);
  const _Float16* ap[MT];
  const _Float16* bp[NT];
#pragma unroll
  for (int m = 0; m < MT; ++m) ap[m] = A + (size_t)(rm0 + m * 16 + l15) * K;
#pragma unroll
  for (int n = 0; n < NT; ++n) bp[n] = Bt + (size_t)(cn0 + n * 16 + l15) * K;
  const v8f vzero = {};
  v8f acc[MT][NT];
#pragma unroll
  for (int m = 0; m < MT; ++m)
#pragma unroll
    for (int n = 0; n < NT; ++n) acc[m][n] = vzero;

  for (int kk = 0; kk < K; kk += 32) {
    v16h a[MT], b[NT];
#pragma unroll
    for (int m = 0; m < MT; ++m) {
      v8h lo = *(const v8h*)(ap[m] + kk + kb);
      v8h hi = *(const v8h*)(ap[m] + kk + 16 + kb);
#pragma unroll
      for (int i = 0; i < 8; ++i) { a[m][i] = lo[i]; a[m][8 + i] = hi[i]; }
    }
#pragma unroll
    for (int n = 0; n < NT; ++n) {
      v8h lo = *(const v8h*)(bp[n] + kk + kb);
      v8h hi = *(const v8h*)(bp[n] + kk + 16 + kb);
#pragma unroll
      for (int i = 0; i < 8; ++i) { b[n][i] = lo[i]; b[n][8 + i] = hi[i]; }
    }
#pragma unroll
    for (int m = 0; m < MT; ++m)
#pragma unroll
      for (int n = 0; n < NT; ++n)
        acc[m][n] = __builtin_amdgcn_wmma_f32_16x16x32_f16(
            false, a[m], false, b[n], (short)0, acc[m][n], false, false);
  }
#pragma unroll
  for (int n = 0; n < NT; ++n) {
    const int cn = cn0 + n * 16 + l15;
    const float bv = bias ? bias[cn] : 0.0f;
#pragma unroll
    for (int m = 0; m < MT; ++m) {
      const int rbase = rm0 + m * 16 + ((lane < 16) ? 0 : 8);
#pragma unroll
      for (int i = 0; i < 8; ++i) {
        float v = acc[m][n][i] + bv;
        if (relu) v = fmaxf(v, 0.0f);
        if (Cf) Cf[(size_t)(rbase + i) * N + cn] = v;
        if (Ch) Ch[(size_t)(rbase + i) * N + cn] = (_Float16)v;
      }
    }
  }
}

// ---------------- layernorm over rows of 256 (optional residual) ------------
__global__ void ln_rows(float* __restrict__ out, const float* __restrict__ a,
                        const float* __restrict__ b, const float* __restrict__ g,
                        const float* __restrict__ beta) {
  __shared__ float red[256];
  const int r = blockIdx.x, c = threadIdx.x;
  float x = a[(size_t)r * D + c] + (b ? b[(size_t)r * D + c] : 0.f);
  red[c] = x; __syncthreads();
  for (int s = 128; s > 0; s >>= 1) { if (c < s) red[c] += red[c + s]; __syncthreads(); }
  const float m = red[0] * (1.0f / D); __syncthreads();
  const float dx = x - m;
  red[c] = dx * dx; __syncthreads();
  for (int s = 128; s > 0; s >>= 1) { if (c < s) red[c] += red[c + s]; __syncthreads(); }
  const float var = red[0] * (1.0f / D);
  out[(size_t)r * D + c] = dx * rsqrtf(var + 1e-5f) * g[c] + beta[c];
}

// ---------------- qmask: qm[n,t,p] = sum_c qn[n,c] * src[c,t,p] -------------
__global__ void qmask_k(const float* __restrict__ qn, const float* __restrict__ src,
                        float* __restrict__ qm, int hw) {
  const long i = (long)blockIdx.x * 256 + threadIdx.x;
  const long tot = (long)NQ * T * hw;
  if (i >= tot) return;
  const int p = (int)(i % hw), t = (int)((i / hw) % T), n = (int)(i / ((long)hw * T));
  float s = 0.f;
  for (int c = 0; c < D; ++c) s += qn[n * D + c] * src[((size_t)c * T + t) * hw + p];
  qm[i] = s;
}

// ---------------- exact top-k via rank count (matches lax.top_k + reverse) --
__global__ void topk_k(const float* __restrict__ qm, int* __restrict__ idx,
                       int hw, int k) {
  extern __shared__ float vals[];
  const int b = blockIdx.x;                    // n*T + t
  const float* row = qm + (size_t)b * hw;
  for (int p = threadIdx.x; p < hw; p += 256) vals[p] = row[p];
  __syncthreads();
  for (int p = threadIdx.x; p < hw; p += 256) {
    const float v = vals[p];
    int rank = 0;
    for (int j = 0; j < hw; ++j) {
      const float u = vals[j];
      rank += (u > v) || (u == v && j < p);
    }
    if (rank < k) idx[(size_t)b * k + (k - 1 - rank)] = p;  // reversed order
  }
}

// ---------------- depthwise 3x3x3 conv over (t,h,w), pad 1 ------------------
__global__ void dwconv3d_k(const float* __restrict__ src, const float* __restrict__ w,
                           const float* __restrict__ bias, float* __restrict__ out,
                           int H, int W) {
  const long i = (long)blockIdx.x * 256 + threadIdx.x;
  const long tot = (long)D * T * H * W;
  if (i >= tot) return;
  const int x = (int)(i % W), y = (int)((i / W) % H);
  const int t = (int)((i / ((long)W * H)) % T), c = (int)(i / ((long)W * H * T));
  float s = bias[c];
  for (int dz = 0; dz < 3; ++dz) {
    const int tt = t + dz - 1; if (tt < 0 || tt >= T) continue;
    for (int dy = 0; dy < 3; ++dy) {
      const int yy = y + dy - 1; if (yy < 0 || yy >= H) continue;
      for (int dx = 0; dx < 3; ++dx) {
        const int xx = x + dx - 1; if (xx < 0 || xx >= W) continue;
        s += w[c * 27 + dz * 9 + dy * 3 + dx] *
             src[(((size_t)c * T + tt) * H + yy) * W + xx];
      }
    }
  }
  out[i] = s;
}

// ---------------- build sequence (nq, d, 1020) ------------------------------
__global__ void build_seq_k(const float* __restrict__ q,
                            const float* __restrict__ v0, const float* __restrict__ v1,
                            const float* __restrict__ v2,
                            const int* __restrict__ i0, const int* __restrict__ i1,
                            const int* __restrict__ i2, float* __restrict__ x) {
  const long i = (long)blockIdx.x * 256 + threadIdx.x;
  const long tot = (long)NQ * D * LTOT;
  if (i >= tot) return;
  const int pos = (int)(i % LTOT), c = (int)((i / LTOT) % D), n = (int)(i / ((long)LTOT * D));
  const int t = pos / LSEG, slot = pos % LSEG;
  float val;
  if (slot < 256)        val = v0[((size_t)c * T + t) * HW0 + i0[(n * T + t) * K0 + slot]];
  else if (slot == 256)  val = q[n * D + c];
  else if (slot < 321)   val = v1[((size_t)c * T + t) * HW1 + i1[(n * T + t) * K1 + (slot - 257)]];
  else if (slot == 321)  val = q[n * D + c];
  else if (slot < 338)   val = v2[((size_t)c * T + t) * HW2 + i2[(n * T + t) * K2 + (slot - 322)]];
  else                   val = q[n * D + c];
  x[i] = val;
}

// ---------------- causal depthwise conv1d (D_CONV=4) + silu ----------------
__global__ void conv1d_silu_k(const float* __restrict__ x, const float* __restrict__ cw,
                              const float* __restrict__ cb, float* __restrict__ xs) {
  const long i = (long)blockIdx.x * 256 + threadIdx.x;
  const long tot = (long)NQ * D * LTOT;
  if (i >= tot) return;
  const int l = (int)(i % LTOT), c = (int)((i / LTOT) % D);
  const long base = i - l;
  float s = cb[c];
  for (int j = 0; j < 4; ++j) {
    const int ll = l - 3 + j;
    if (ll >= 0) s += cw[c * 4 + j] * x[base + ll];
  }
  xs[i] = s / (1.f + expf(-s));                 // silu
}

// ---------------- transpose (q,c,l) -> (token,c) f16, pad rows --------------
__global__ void cvt_tok_k(const float* __restrict__ xs, _Float16* __restrict__ xt) {
  const long i = (long)blockIdx.x * 256 + threadIdx.x;
  const long tot = (long)NTOKP * D;
  if (i >= tot) return;
  const int c = (int)(i % D); const long tok = i / D;
  if (tok < NTOK) {
    const int n = (int)(tok / LTOT), l = (int)(tok % LTOT);
    xt[i] = (_Float16)xs[((size_t)n * D + c) * LTOT + l];
  } else xt[i] = (_Float16)0.f;
}

// ---------------- weight transpose + f16: Wt[n,k] = W[k,n] ------------------
__global__ void trcvt_k(const float* __restrict__ W, _Float16* __restrict__ Wt,
                        int K, int N, int realN) {
  const long i = (long)blockIdx.x * 256 + threadIdx.x;
  if (i >= (long)N * K) return;
  const int k = (int)(i % K), n = (int)(i / K);
  Wt[i] = (n < realN) ? (_Float16)W[(size_t)k * realN + n] : (_Float16)0.f;
}

// ---------------- f32 -> f16, row-padded ------------------------------------
__global__ void padcvt_k(const float* __restrict__ src, _Float16* __restrict__ dst,
                         int rows, int padrows, int cols) {
  const long i = (long)blockIdx.x * 256 + threadIdx.x;
  if (i >= (long)padrows * cols) return;
  const int r = (int)(i / cols);
  dst[i] = (r < rows) ? (_Float16)src[i] : (_Float16)0.f;
}

// ---------------- S6 selective scan: one block per query --------------------
#define CHK 32
__global__ void s6_scan_k(const float* __restrict__ xs, const float* __restrict__ dbl,
                          const float* __restrict__ dtw, const float* __restrict__ dtb,
                          const float* __restrict__ Alog, const float* __restrict__ Dp,
                          float* __restrict__ y) {
  __shared__ float xbuf[D][CHK + 1];            // +1 pad to kill bank conflicts
  __shared__ float dbuf[CHK][40];
  const int nqi = blockIdx.x, c = threadIdx.x;
  float w16[16];
#pragma unroll
  for (int j = 0; j < 16; ++j) w16[j] = dtw[j * D + c];
  float Ar[8];
#pragma unroll
  for (int n = 0; n < 8; ++n) Ar[n] = -expf(Alog[c * 8 + n]);
  const float dtbc = dtb[c], dpc = Dp[c];
  float h[8] = {0, 0, 0, 0, 0, 0, 0, 0};
  const size_t xb = ((size_t)nqi * D + c) * LTOT;
  for (int lc = 0; lc < LTOT; lc += CHK) {
    const int cnt = (LTOT - lc < CHK) ? (LTOT - lc) : CHK;
    for (int i = 0; i < cnt; ++i) xbuf[c][i] = xs[xb + lc + i];
    for (int i = c; i < cnt * 40; i += D) {
      const int li = i / 40, jj = i % 40;
      dbuf[li][jj] = dbl[((size_t)nqi * LTOT + lc + li) * 48 + jj];
    }
    __syncthreads();
    for (int i = 0; i < cnt; ++i) {
      float acc = dtbc;
#pragma unroll
      for (int j = 0; j < 16; ++j) acc += dbuf[i][j] * w16[j];
      const float dt = (acc > 20.f) ? acc : log1pf(expf(acc));   // softplus
      const float xv = xbuf[c][i];
      float yv = 0.f;
#pragma unroll
      for (int n = 0; n < 8; ++n) {
        const float a  = expf(dt * Ar[n]);
        const float bx = dt * xv * dbuf[i][16 + n];
        h[n] = a * h[n] + bx;
        yv += h[n] * dbuf[i][24 + n];
      }
      y[xb + lc + i] = yv + xv * dpc;
    }
    __syncthreads();
  }
}

// ---------------- q_out pre-attention ---------------------------------------
__global__ void qpre_k(const float* __restrict__ q, const float* __restrict__ y,
                       float* __restrict__ qpre) {
  const int i = blockIdx.x * 256 + threadIdx.x;
  if (i >= NQ * D) return;
  const size_t base = (size_t)i * LTOT;   // i = n*D + c, y layout (n,c,l)
  float tm = 0.f, ms = 0.f;
  for (int t = 0; t < T; ++t) {
    tm += y[base + t * LSEG + 339];
    ms += y[base + t * LSEG + 256] + y[base + t * LSEG + 321] + y[base + t * LSEG + 338];
  }
  qpre[i] = q[i] + tm * (1.f / 3.f) + ms * (1.f / 9.f);
}

// ---------------- ori: S[t,p,c] = src_s[c,t,pp] -----------------------------
__global__ void ori_k(const float* __restrict__ s0, const float* __restrict__ s1,
                      const float* __restrict__ s2, float* __restrict__ S) {
  const long i = (long)blockIdx.x * 256 + threadIdx.x;
  if (i >= (long)T * HWALL * D) return;
  const int c = (int)(i % D), p = (int)((i / D) % HWALL), t = (int)(i / ((long)D * HWALL));
  float v;
  if (p < HW0)            v = s0[((size_t)c * T + t) * HW0 + p];
  else if (p < HW0 + HW1) v = s1[((size_t)c * T + t) * HW1 + (p - HW0)];
  else                    v = s2[((size_t)c * T + t) * HW2 + (p - HW0 - HW1)];
  S[i] = v;
}

// ---------------- scatter-add gathered features back ------------------------
__global__ void scatter_k(const float* __restrict__ y, const int* __restrict__ idx,
                          float* __restrict__ S, int k, int offc, int basep) {
  const long i = (long)blockIdx.x * 256 + threadIdx.x;
  const long tot = (long)T * NQ * k * D;
  if (i >= tot) return;
  const int c = (int)(i % D), j = (int)((i / D) % k);
  const int n = (int)((i / ((long)D * k)) % NQ), t = (int)(i / ((long)D * k * NQ));
  const float val = y[((size_t)n * D + c) * LTOT + t * LSEG + offc + j];
  const int row = basep + idx[(n * T + t) * k + j];
  atomicAdd(&S[((size_t)t * HWALL + row) * D + c], val);
}

// ---------------- 8-head attention over 25 queries --------------------------
__global__ void attn_k(const float* __restrict__ qkv, float* __restrict__ o) {
  __shared__ float sc[32];
  const int q = blockIdx.x, h = blockIdx.y, lane = threadIdx.x;
  float s = -1e30f;
  if (lane < NQ) {
    float acc = 0.f;
    for (int dd = 0; dd < 32; ++dd)
      acc += qkv[q * 768 + h * 32 + dd] * qkv[lane * 768 + 256 + h * 32 + dd];
    s = acc * 0.17677669529663687f;             // 1/sqrt(32)
  }
  sc[lane] = s; __syncthreads();
  float mx = -1e30f;
  for (int k = 0; k < NQ; ++k) mx = fmaxf(mx, sc[k]);
  const float e = (lane < NQ) ? expf(s - mx) : 0.f;
  __syncthreads();
  sc[lane] = e; __syncthreads();
  float sum = 0.f;
  for (int k = 0; k < NQ; ++k) sum += sc[k];
  const float inv = 1.f / sum;
  float acc = 0.f;                              // lane = output dim (dh = 32)
  for (int k = 0; k < NQ; ++k) acc += sc[k] * qkv[k * 768 + 512 + h * 32 + lane];
  o[q * D + h * 32 + lane] = acc * inv;
}

// ============================================================================
extern "C" void kernel_launch(void* const* d_in, const int* in_sizes, int n_in,
                              void* d_out, int out_size, void* d_ws, size_t ws_size,
                              hipStream_t stream) {
  // inputs in setup_inputs() dict insertion order
  const float* src0 = (const float*)d_in[0];
  const float* src1 = (const float*)d_in[1];
  const float* src2 = (const float*)d_in[2];
  const float* qry  = (const float*)d_in[3];
  const float* qn_g = (const float*)d_in[4];   const float* qn_b = (const float*)d_in[5];
  const float* vp_w = (const float*)d_in[6];   const float* vp_b = (const float*)d_in[7];
  const float* cw   = (const float*)d_in[8];   const float* cb   = (const float*)d_in[9];
  const float* x_proj = (const float*)d_in[10];
  const float* dt_w = (const float*)d_in[11];  const float* dt_b = (const float*)d_in[12];
  const float* A_log = (const float*)d_in[13]; const float* Dpv  = (const float*)d_in[14];
  const float* n1_g = (const float*)d_in[15];  const float* n1_b = (const float*)d_in[16];
  const float* ff1_w = (const float*)d_in[17]; const float* ff1_b = (const float*)d_in[18];
  const float* ff2_w = (const float*)d_in[19]; const float* ff2_b = (const float*)d_in[20];
  const float* ffn_g = (const float*)d_in[21]; const float* ffn_b = (const float*)d_in[22];
  const float* wqkv = (const float*)d_in[23];  const float* bqkv = (const float*)d_in[24];
  const float* wo   = (const float*)d_in[25];  const float* bo   = (const float*)d_in[26];
  const float* sa_g = (const float*)d_in[27];  const float* sa_b = (const float*)d_in[28];
  const float* qf1_w = (const float*)d_in[29]; const float* qf1_b = (const float*)d_in[30];
  const float* qf2_w = (const float*)d_in[31]; const float* qf2_b = (const float*)d_in[32];
  const float* qfn_g = (const float*)d_in[33]; const float* qfn_b = (const float*)d_in[34];

  size_t off = 0;
  auto alloc = [&](size_t bytes) -> void* {
    off = (off + 255) & ~(size_t)255;
    void* p = (char*)d_ws + off;
    off += bytes;
    return p;
  };
  float* qn   = (float*)alloc((size_t)NQ * D * 4);
  float* qm0  = (float*)alloc((size_t)NQ * T * HW0 * 4);
  float* qm1  = (float*)alloc((size_t)NQ * T * HW1 * 4);
  float* qm2  = (float*)alloc((size_t)NQ * T * HW2 * 4);
  int*   i0   = (int*)alloc((size_t)NQ * T * K0 * 4);
  int*   i1   = (int*)alloc((size_t)NQ * T * K1 * 4);
  int*   i2   = (int*)alloc((size_t)NQ * T * K2 * 4);
  float* v0   = (float*)alloc((size_t)D * T * HW0 * 4);
  float* v1   = (float*)alloc((size_t)D * T * HW1 * 4);
  float* v2   = (float*)alloc((size_t)D * T * HW2 * 4);
  float* xseq = (float*)alloc((size_t)NQ * D * LTOT * 4);
  float* xsb  = (float*)alloc((size_t)NQ * D * LTOT * 4);
  _Float16* xt16   = (_Float16*)alloc((size_t)NTOKP * D * 2);
  _Float16* xprojT = (_Float16*)alloc((size_t)48 * D * 2);
  float* dbl  = (float*)alloc((size_t)NTOKP * 48 * 4);
  float* yb   = xseq;                           // alias: xseq dead after conv1d
  float* qpre = (float*)alloc((size_t)NQ * D * 4);
  float* S    = (float*)alloc((size_t)ROWS * D * 4);
  float* srcln = (float*)alloc((size_t)ROWS * D * 4);
  _Float16* X16 = (_Float16*)alloc((size_t)ROWS * D * 2);
  _Float16* W1t = (_Float16*)alloc((size_t)DFF * D * 2);
  _Float16* W2t = (_Float16*)alloc((size_t)D * DFF * 2);
  _Float16* H16 = (_Float16*)alloc((size_t)ROWS * DFF * 2);
  float* Yb   = (float*)alloc((size_t)ROWS * D * 4);
  _Float16* WqkvT = (_Float16*)alloc((size_t)768 * D * 2);
  _Float16* WoT   = (_Float16*)alloc((size_t)D * D * 2);
  _Float16* Wq1t  = (_Float16*)alloc((size_t)DFF * D * 2);
  _Float16* Wq2t  = (_Float16*)alloc((size_t)D * DFF * 2);
  _Float16* qpad  = (_Float16*)alloc((size_t)32 * D * 2);
  float* qkvb = (float*)alloc((size_t)32 * 768 * 4);
  float* ob   = (float*)alloc((size_t)NQ * D * 4);
  _Float16* o16 = (_Float16*)alloc((size_t)32 * D * 2);
  float* O2   = (float*)alloc((size_t)32 * D * 4);
  float* q2   = (float*)alloc((size_t)NQ * D * 4);
  _Float16* q2p = (_Float16*)alloc((size_t)32 * D * 2);
  _Float16* Hq16 = (_Float16*)alloc((size_t)32 * DFF * 2);
  float* Yq   = (float*)alloc((size_t)32 * D * 4);
  (void)ws_size; (void)n_in; (void)in_sizes; (void)out_size;

  float* out_src = (float*)d_out;               // (3*1344, 256)
  float* out_q   = (float*)d_out + (size_t)ROWS * D;

  auto g = [](long n) { return (unsigned)((n + 255) / 256); };

  // 1) qn = LN(queries)
  ln_rows<<<NQ, 256, 0, stream>>>(qn, qry, nullptr, qn_g, qn_b);
  // 2) query masks
  qmask_k<<<g((long)NQ * T * HW0), 256, 0, stream>>>(qn, src0, qm0, HW0);
  qmask_k<<<g((long)NQ * T * HW1), 256, 0, stream>>>(qn, src1, qm1, HW1);
  qmask_k<<<g((long)NQ * T * HW2), 256, 0, stream>>>(qn, src2, qm2, HW2);
  // 3) top-k (reversed)
  topk_k<<<NQ * T, 256, HW0 * 4, stream>>>(qm0, i0, HW0, K0);
  topk_k<<<NQ * T, 256, HW1 * 4, stream>>>(qm1, i1, HW1, K1);
  topk_k<<<NQ * T, 256, HW2 * 4, stream>>>(qm2, i2, HW2, K2);
  // 4) depthwise 3D value convs
  dwconv3d_k<<<g((long)D * T * HW0), 256, 0, stream>>>(src0, vp_w, vp_b, v0, 32, 32);
  dwconv3d_k<<<g((long)D * T * HW1), 256, 0, stream>>>(src1, vp_w, vp_b, v1, 16, 16);
  dwconv3d_k<<<g((long)D * T * HW2), 256, 0, stream>>>(src2, vp_w, vp_b, v2, 8, 8);
  // 5) weight transposes to f16 (N x K) for WMMA B operands
  trcvt_k<<<g((long)48 * D), 256, 0, stream>>>(x_proj, xprojT, D, 48, 40);
  trcvt_k<<<g((long)DFF * D), 256, 0, stream>>>(ff1_w, W1t, D, DFF, DFF);
  trcvt_k<<<g((long)D * DFF), 256, 0, stream>>>(ff2_w, W2t, DFF, D, D);
  trcvt_k<<<g((long)768 * D), 256, 0, stream>>>(wqkv, WqkvT, D, 768, 768);
  trcvt_k<<<g((long)D * D), 256, 0, stream>>>(wo, WoT, D, D, D);
  trcvt_k<<<g((long)DFF * D), 256, 0, stream>>>(qf1_w, Wq1t, D, DFF, DFF);
  trcvt_k<<<g((long)D * DFF), 256, 0, stream>>>(qf2_w, Wq2t, DFF, D, D);
  // 6) gather -> sequence, 7) causal conv + silu
  build_seq_k<<<g((long)NQ * D * LTOT), 256, 0, stream>>>(qry, v0, v1, v2, i0, i1, i2, xseq);
  conv1d_silu_k<<<g((long)NQ * D * LTOT), 256, 0, stream>>>(xseq, cw, cb, xsb);
  // 8) tokens -> f16, 9) x_proj via WMMA (M=25504, N=48 -> 2x3 tile block)
  cvt_tok_k<<<g((long)NTOKP * D), 256, 0, stream>>>(xsb, xt16);
  wmma_gemm_t<2, 3><<<dim3(NTOKP / 32, 1), 32, 0, stream>>>(
      xt16, xprojT, nullptr, dbl, nullptr, NTOKP, 48, D, 0);
  // 10) S6 scan (one block per query, LDS-staged dbl + x)
  s6_scan_k<<<NQ, 256, 0, stream>>>(xsb, dbl, dt_w, dt_b, A_log, Dpv, yb);
  // 11) query pre-attention mix
  qpre_k<<<g((long)NQ * D), 256, 0, stream>>>(qry, yb, qpre);
  // 12) ori + 13) scatter-add
  ori_k<<<g((long)ROWS * D), 256, 0, stream>>>(src0, src1, src2, S);
  scatter_k<<<g((long)T * NQ * K0 * D), 256, 0, stream>>>(yb, i0, S, K0, 0, 0);
  scatter_k<<<g((long)T * NQ * K1 * D), 256, 0, stream>>>(yb, i1, S, K1, 257, HW0);
  scatter_k<<<g((long)T * NQ * K2 * D), 256, 0, stream>>>(yb, i2, S, K2, 322, HW0 + HW1);
  // 14) src LN, 15-17) FFN via 2x4-blocked WMMA, 18) final src LN -> d_out
  ln_rows<<<ROWS, 256, 0, stream>>>(srcln, S, nullptr, n1_g, n1_b);
  padcvt_k<<<g((long)ROWS * D), 256, 0, stream>>>(srcln, X16, ROWS, ROWS, D);
  wmma_gemm_t<2, 4><<<dim3(ROWS / 32, DFF / 64), 32, 0, stream>>>(
      X16, W1t, ff1_b, nullptr, H16, ROWS, DFF, D, 1);
  wmma_gemm_t<2, 4><<<dim3(ROWS / 32, D / 64), 32, 0, stream>>>(
      H16, W2t, ff2_b, Yb, nullptr, ROWS, D, DFF, 0);
  ln_rows<<<ROWS, 256, 0, stream>>>(out_src, srcln, Yb, ffn_g, ffn_b);
  // 19-23) query attention block via WMMA (M padded 25->32)
  padcvt_k<<<g((long)32 * D), 256, 0, stream>>>(qpre, qpad, NQ, 32, D);
  wmma_gemm_t<2, 4><<<dim3(1, 768 / 64), 32, 0, stream>>>(
      qpad, WqkvT, bqkv, qkvb, nullptr, 32, 768, D, 0);
  attn_k<<<dim3(NQ, 8), 32, 0, stream>>>(qkvb, ob);
  padcvt_k<<<g((long)32 * D), 256, 0, stream>>>(ob, o16, NQ, 32, D);
  wmma_gemm_t<2, 4><<<dim3(1, D / 64), 32, 0, stream>>>(
      o16, WoT, bo, O2, nullptr, 32, D, D, 0);
  ln_rows<<<NQ, 256, 0, stream>>>(q2, qpre, O2, sa_g, sa_b);
  // 25-28) query FFN + final LN -> d_out tail
  padcvt_k<<<g((long)32 * D), 256, 0, stream>>>(q2, q2p, NQ, 32, D);
  wmma_gemm_t<2, 4><<<dim3(1, DFF / 64), 32, 0, stream>>>(
      q2p, Wq1t, qf1_b, nullptr, Hq16, 32, DFF, D, 1);
  wmma_gemm_t<2, 4><<<dim3(1, D / 64), 32, 0, stream>>>(
      Hq16, Wq2t, qf2_b, Yq, nullptr, 32, D, DFF, 0);
  ln_rows<<<NQ, 256, 0, stream>>>(out_q, q2, Yq, qfn_g, qfn_b);
}
